// MyConvBlock_55585466744988
// MI455X (gfx1250) — matrix operand order repaired and checked by reading, not compiled, not run on hardware
//
#include <hip/hip_runtime.h>

typedef __attribute__((ext_vector_type(16))) _Float16 v16h;
typedef __attribute__((ext_vector_type(8)))  float    v8f;

#define BS    8
#define CIN   64
#define COUT  128
#define NPIX  256          // input H = W
#define HO    254
#define WO    254
#define KTOT  576          // CIN * 9
#define NKSTEP 18          // KTOT / 32
#define MTILES 8           // COUT / 16

#define WSWZ_HALFS (MTILES * NKSTEP * 32 * 16)   // 73728 fp16 = 144 KB (A-fragment order)
#define XP_STRIDE  592                           // panel row stride in halves (32B aligned)
#define XPAN_HALFS (128 * XP_STRIDE)             // ~148 KB

// ---- A-fragment (16x32 f16) swizzle math, contiguous k order, same as the B panel reads:
// lane l -> M = l&15 ; element e (vg = e>>1):
//   k_in_frag = (l>>4)*16 + e
__device__ __forceinline__ float swz_weight_fetch(const float* __restrict__ w, int idx) {
    const int frag = idx >> 9;          // s*8 + t  (512 halves per fragment)
    const int rem  = idx & 511;
    const int l    = rem >> 4;
    const int e    = rem & 15;
    const int s    = frag >> 3;
    const int t    = frag & 7;
    const int vg   = e >> 1;
    const int kin  = ((l >> 4) * 16) + e;
    const int k    = s * 32 + kin;
    const int m    = t * 16 + (l & 15);
    const int c    = k / 9;
    const int r9   = k - c * 9;
    const int kh   = r9 / 3;
    const int kw   = r9 - kh * 3;
    return w[((m * CIN + c) * 3 + kh) * 3 + kw];
}

// Kernel 1: one-time weight swizzle f32 -> fp16 A-fragment order in global scratch.
// 144 KB result stays L2-resident for the whole conv.
__global__ void swizzle_weights(const float* __restrict__ w, _Float16* __restrict__ wswz) {
    const int idx = blockIdx.x * 256 + threadIdx.x;
    if (idx < WSWZ_HALFS) wswz[idx] = (_Float16)swz_weight_fetch(w, idx);
}

// Kernel 2: implicit-GEMM conv. Block = 128-pixel strip of one output row, all 128 channels.
// USE_WS=true : A fragments streamed from L2-resident d_ws (global_load_b128),
//               LDS holds only the x panel -> 2 blocks/WGP.
// USE_WS=false: fallback, weights swizzled into LDS in-kernel -> 1 block/WGP.
template <bool USE_WS>
__global__ __launch_bounds__(256, USE_WS ? 2 : 1)
void conv3x3_wmma_f16(const float* __restrict__ x,
                      const float* __restrict__ w,
                      const _Float16* __restrict__ wswz,
                      float* __restrict__ out)
{
    extern __shared__ _Float16 smem[];
    _Float16* xpan = USE_WS ? smem : (smem + WSWZ_HALFS);  // im2col panel [pixel][k]

    const int tid   = threadIdx.x;
    const int bid   = blockIdx.x;
    const int strip = bid & 1;
    const int h     = (bid >> 1) % HO;
    const int b     = bid / (2 * HO);
    const int w0    = strip * 128;
    const int npix  = strip ? (WO - 128) : 128;   // 126 or 128

    if constexpr (!USE_WS) {
        _Float16* wlds = smem;
        for (int idx = tid; idx < WSWZ_HALFS; idx += 256)
            wlds[idx] = (_Float16)swz_weight_fetch(w, idx);
    }

    // ---- build im2col panel: panel[p][c*9+kh*3+kw] = x[b][c][h+kh][w0+p+kw]
    // Value-centric: each coalesced global read scatters to up to 3 panel slots.
    for (int idx = tid; idx < CIN * 3 * 130; idx += 256) {
        const int c  = idx / 390;
        const int r  = idx - c * 390;
        const int kh = r / 130;
        const int j  = r - kh * 130;                  // 0..129
        int col = w0 + j;
        if (col > NPIX - 1) col = NPIX - 1;           // only feeds masked pixels
        const _Float16 xv = (_Float16)x[((b * CIN + c) * NPIX + (h + kh)) * NPIX + col];
        const int kb = c * 9 + kh * 3;
        #pragma unroll
        for (int kw = 0; kw < 3; ++kw) {
            const int p = j - kw;
            if (p >= 0 && p < 128) xpan[p * XP_STRIDE + kb + kw] = xv;
        }
    }
    __syncthreads();

    const int lane  = tid & 31;
    const int wave  = tid >> 5;
    const int nlo   = lane & 15;     // B/D column within 16-wide tile
    const int khalf = lane >> 4;     // B half: k+0..15 vs k+16..31 ; D rows 0..7 vs 8..15
    const int mw    = wave & 3;      // -> M-tiles 2*mw, 2*mw+1
    const int nw    = wave >> 2;     // -> pixel groups nw*4 .. nw*4+3

    v8f acc[2][4];
    #pragma unroll
    for (int i = 0; i < 2; ++i)
        #pragma unroll
        for (int g = 0; g < 4; ++g)
            acc[i][g] = (v8f){0.f, 0.f, 0.f, 0.f, 0.f, 0.f, 0.f, 0.f};

    // A-fragment source: global (L2-resident) or LDS fallback
    const _Float16* afsrc;
    if constexpr (USE_WS) afsrc = wswz; else afsrc = smem;
    const v16h* af = ((const v16h*)afsrc) + (2 * mw) * 32 + lane;

    // per-lane B row bases (pixel rows of the panel), contiguous v16h per k-step
    const _Float16* brow[4];
    #pragma unroll
    for (int g = 0; g < 4; ++g)
        brow[g] = xpan + (nw * 64 + g * 16 + nlo) * XP_STRIDE + khalf * 16;

    for (int s = 0; s < NKSTEP; ++s) {
        const v16h a0 = af[s * (MTILES * 32) + 0];    // 2x {global|ds}_load_b128, coalesced
        const v16h a1 = af[s * (MTILES * 32) + 32];
        const int coff = s * 32;
        #pragma unroll
        for (int g = 0; g < 4; ++g) {
            const v16h bf = *(const v16h*)(brow[g] + coff);   // 2x ds_load_b128
            acc[0][g] = __builtin_amdgcn_wmma_f32_16x16x32_f16(
                false, a0, false, bf, (short)0, acc[0][g], false, false);
            acc[1][g] = __builtin_amdgcn_wmma_f32_16x16x32_f16(
                false, a1, false, bf, (short)0, acc[1][g], false, false);
        }
    }

    // ---- store D tiles: lane -> N = lane&15 ; VGPR r -> M = khalf*8 + r
    #pragma unroll
    for (int g = 0; g < 4; ++g) {
        const int pcol = nw * 64 + g * 16 + nlo;
        if (pcol < npix) {
            const int ow = w0 + pcol;
            #pragma unroll
            for (int i = 0; i < 2; ++i) {
                const int chb = (2 * mw + i) * 16 + khalf * 8;
                #pragma unroll
                for (int r = 0; r < 8; ++r)
                    out[((b * COUT + chb + r) * HO + h) * WO + ow] = acc[i][g][r];
            }
        }
    }
}

extern "C" void kernel_launch(void* const* d_in, const int* in_sizes, int n_in,
                              void* d_out, int out_size, void* d_ws, size_t ws_size,
                              hipStream_t stream) {
    (void)in_sizes; (void)n_in; (void)out_size;
    const float* x = (const float*)d_in[0];
    const float* w = (const float*)d_in[1];
    float* out = (float*)d_out;

    const int grid = BS * HO * 2;   // one 128-pixel strip of one output row per block

    if (ws_size >= (size_t)WSWZ_HALFS * sizeof(_Float16)) {
        _Float16* wswz = (_Float16*)d_ws;
        swizzle_weights<<<(WSWZ_HALFS + 255) / 256, 256, 0, stream>>>(w, wswz);
        conv3x3_wmma_f16<true><<<grid, 256, XPAN_HALFS * 2, stream>>>(x, w, wswz, out);
    } else {
        conv3x3_wmma_f16<false><<<grid, 256, (WSWZ_HALFS + XPAN_HALFS) * 2, stream>>>(
            x, w, nullptr, out);
    }
}
